// KAN_v0_455266533466
// MI455X (gfx1250) — matrix operand order repaired and checked by reading, not compiled, not run on hardware
//
#include <hip/hip_runtime.h>
#include <hip/hip_bf16.h>

typedef __attribute__((ext_vector_type(16))) _Float16 v16h;
typedef __attribute__((ext_vector_type(8)))  _Float16 v8h;
typedef __attribute__((ext_vector_type(8)))  float    v8f;

#define IMG_H 384
#define IMG_W 384
#define HW (IMG_H * IMG_W)          // 147456
#define NPIX (8 * HW)               // 1179648
#define NCONV 16
#define KFEAT 96                    // 81 real features padded to 96 (3 x K=32 WMMA steps)
#define FSTRIDE 104                 // LDS row stride in halves: 52 words, conflict-free, 16B aligned
#define WAVES 8

// Feature layout (and matching weight-row layout):
//   rows 0..71  : spline bases, element k -> slots 8k..8k+7 (16B-aligned blocks)
//   rows 72..80 : SiLU(patch) features
//   rows 81..95 : zero padding (zeroed: 0 * NaN would poison WMMA)
//   slots 96..103: scrap area — scatter stores with out-of-range basis index
//                  land here unconditionally; never read by WMMA fragments.

__device__ __forceinline__ v16h ld16h_contig(const _Float16* p) {
    v8h lo = *(const v8h*)p;
    v8h hi = *(const v8h*)(p + 8);
    return __builtin_shufflevector(lo, hi, 0,1,2,3,4,5,6,7,8,9,10,11,12,13,14,15);
}

__device__ __forceinline__ v16h ldA_frag(const _Float16* row, int c, int khalf) {
    // 16-bit A 16x32 layout: elems 0..7 -> K = 8*khalf + e ; elems 8..15 -> K = 16 + 8*khalf + (e-8)
    v8h lo = *(const v8h*)(row + c * 32 + khalf * 8);
    v8h hi = *(const v8h*)(row + c * 32 + 16 + khalf * 8);
    return __builtin_shufflevector(lo, hi, 0,1,2,3,4,5,6,7,8,9,10,11,12,13,14,15);
}

// ---------------------------------------------------------------------------
// Kernel 1: per-pixel KAN features -> WMMA GEMM [N,96] x [96,16] -> y [N,16]
// Uniform-knot fast path: only 4 of 8 cubic B-spline bases are nonzero;
// scatter via per-lane-addressed ds_store_b16 with address clamping (no exec
// masking), patch loads via address clamp + cndmask (no guarded loads).
// ---------------------------------------------------------------------------
__global__ __launch_bounds__(256) void kan_feat_wmma(
    const float* __restrict__ x,          // [8,1,384,384]
    const float* __restrict__ base_w,     // [16,9]
    const float* __restrict__ spline_w,   // [16,9,8]
    const float* __restrict__ spline_s,   // [16,9]
    _Float16* __restrict__ y)             // [NPIX,16]
{
    __shared__ _Float16 sWt[NCONV * KFEAT];          // transposed weights [col][k], 3 KB
    __shared__ _Float16 sF[WAVES][32 * FSTRIDE];     // per-wave feature tiles, 52 KB

    const int tid  = threadIdx.x;
    const int lane = tid & 31;
    const int wave = tid >> 5;

    // Build combined weight matrix, transposed: sWt[c*96 + r]
    for (int idx = tid; idx < NCONV * KFEAT; idx += 256) {
        int c = idx / KFEAT, r = idx - c * KFEAT;
        float w;
        if (r < 72) {
            int k = r >> 3, j = r & 7;
            w = spline_w[(c * 9 + k) * 8 + j] * spline_s[c * 9 + k];
        } else if (r < 81) {
            w = base_w[c * 9 + (r - 72)];
        } else {
            w = 0.0f;
        }
        sWt[idx] = (_Float16)w;
    }

    // ---- per-lane pixel: compute the features into this wave's LDS tile ----
    const int n    = blockIdx.x * 256 + tid;         // NPIX = 4608*256 exactly
    const int bimg = n / HW;
    const int rem  = n - bimg * HW;
    const int yy   = rem / IMG_W;
    const int xx   = rem - yy * IMG_W;
    const float* xb = x + bimg * HW;
    _Float16* frow = &sF[wave][lane * FSTRIDE];

    // Zero the readable 96-wide row first (12 x 16B stores); DS ops from one
    // wave stay in order, so the sparse basis stores below land on zeroed slots.
    {
        v8h z = {};
        #pragma unroll
        for (int r = 0; r < KFEAT; r += 8) *(v8h*)(frow + r) = z;
    }

    #pragma unroll
    for (int kh = 0; kh < 3; ++kh) {
        #pragma unroll
        for (int kw = 0; kw < 3; ++kw) {
            const int k  = kh * 3 + kw;
            const int y2 = yy + kh - 1;
            const int x2 = xx + kw - 1;

            // Branch-free zero-padded patch load: clamp address, select value.
            const bool inb = ((unsigned)y2 < (unsigned)IMG_H) &&
                             ((unsigned)x2 < (unsigned)IMG_W);
            const int yc = min(max(y2, 0), IMG_H - 1);
            const int xc = min(max(x2, 0), IMG_W - 1);
            float v = xb[yc * IMG_W + xc];
            v = inb ? v : 0.0f;

            // SiLU branch feature
            frow[72 + k] = (_Float16)(v / (1.0f + __expf(-v)));

            // Uniform cubic B-spline: interval i = floor((v - g0)/h), local t.
            // floor() == the reference's half-open (v >= g_i && v < g_{i+1}).
            const float u  = (v + 2.2f) * 2.5f;      // (v - knot(0)) / h
            const float fi = floorf(u);
            const float t  = u - fi;
            const int   i  = (int)fi;

            const float t2 = t * t;
            const float t3 = t2 * t;
            const float s  = 1.0f - t;
            float P[4];
            P[0] = (s * s * s) * (1.0f / 6.0f);                    // basis j = i-3
            P[1] = 0.5f * t3 - t2 + (2.0f / 3.0f);                 // basis j = i-2
            P[2] = (-3.0f * t3 + 3.0f * t2 + 3.0f * t + 1.0f) * (1.0f / 6.0f); // j = i-1
            P[3] = t3 * (1.0f / 6.0f);                             // basis j = i

            // Branch-free scatter: out-of-range j (also rejects v outside
            // [-2.2, 2.2)) redirects to the scrap slot 96 — one cndmask on the
            // DS address instead of an exec-mask save/restore per store.
            #pragma unroll
            for (int m = 0; m < 4; ++m) {
                const int j    = i - 3 + m;
                const int slot = ((unsigned)j < 8u) ? (k * 8 + j) : KFEAT;
                frow[slot] = (_Float16)P[m];
            }
        }
    }

    __syncthreads();

    // ---- WMMA: two 16x16 output tiles per wave, K = 96 = 3 x 32 ----
    const int col   = lane & 15;    // output channel (B/D column)
    const int khalf = lane >> 4;

    v16h Bm[3];
    #pragma unroll
    for (int c = 0; c < 3; ++c)     // B 32x16: lane holds col, same K order as A (8*khalf + e, 16 + 8*khalf + e-8)
        Bm[c] = ldA_frag(&sWt[col * KFEAT], c, khalf);

    const int tileBase = blockIdx.x * 256 + wave * 32;
    #pragma unroll
    for (int t = 0; t < 2; ++t) {
        const int m = lane & 15;    // A row within tile
        v8f acc = {0.f, 0.f, 0.f, 0.f, 0.f, 0.f, 0.f, 0.f};
        #pragma unroll
        for (int c = 0; c < 3; ++c) {
            v16h A = ldA_frag(&sF[wave][(t * 16 + m) * FSTRIDE], c, khalf);
            acc = __builtin_amdgcn_wmma_f32_16x16x32_f16(
                false, A, false, Bm[c], (short)0, acc, false, false);
        }
        // D layout: lane -> col = lane&15, VGPR r -> row 8*khalf + r
        #pragma unroll
        for (int r = 0; r < 8; ++r) {
            const int pix = tileBase + t * 16 + khalf * 8 + r;
            y[pix * NCONV + col] = (_Float16)acc[r];
        }
    }
}

// ---------------------------------------------------------------------------
// Kernel 2: restore Conv2d(16 -> 1, 3x3, pad 1) + bias. y is L2-resident f16.
// ---------------------------------------------------------------------------
__global__ __launch_bounds__(256) void kan_restore_conv(
    const _Float16* __restrict__ y,       // [NPIX,16]
    const float* __restrict__ rw,         // [1,16,3,3]
    const float* __restrict__ rb,         // [1]
    float* __restrict__ out)              // [8,1,384,384]
{
    __shared__ float lw[NCONV * 9];
    __shared__ float lb;
    const int tid = threadIdx.x;
    if (tid < NCONV * 9) lw[tid] = rw[tid];
    if (tid == NCONV * 9) lb = rb[0];
    __syncthreads();

    const int n    = blockIdx.x * 256 + tid;
    const int bimg = n / HW;
    const int rem  = n - bimg * HW;
    const int yy   = rem / IMG_W;
    const int xx   = rem - yy * IMG_W;

    float acc = lb;
    #pragma unroll
    for (int kh = 0; kh < 3; ++kh) {
        const int y2 = yy + kh - 1;
        if ((unsigned)y2 >= (unsigned)IMG_H) continue;
        #pragma unroll
        for (int kw = 0; kw < 3; ++kw) {
            const int x2 = xx + kw - 1;
            if ((unsigned)x2 >= (unsigned)IMG_W) continue;
            const v16h yv = *(const v16h*)(y + (size_t)(bimg * HW + y2 * IMG_W + x2) * NCONV);
            #pragma unroll
            for (int c = 0; c < NCONV; ++c)
                acc = fmaf((float)yv[c], lw[c * 9 + kh * 3 + kw], acc);
        }
    }
    out[n] = acc;
}

extern "C" void kernel_launch(void* const* d_in, const int* in_sizes, int n_in,
                              void* d_out, int out_size, void* d_ws, size_t ws_size,
                              hipStream_t stream) {
    const float* x        = (const float*)d_in[0];   // [8,1,384,384]
    const float* base_w   = (const float*)d_in[1];   // [16,9]
    const float* spline_w = (const float*)d_in[2];   // [16,9,8]
    const float* spline_s = (const float*)d_in[3];   // [16,9]
    const float* rw       = (const float*)d_in[4];   // [1,16,3,3]
    const float* rb       = (const float*)d_in[5];   // [1]
    float* out            = (float*)d_out;
    _Float16* yws         = (_Float16*)d_ws;         // [NPIX,16] f16 ~= 38 MB (L2-resident)

    const int blocks = NPIX / 256;                   // 4608, exact
    kan_feat_wmma<<<dim3(blocks), dim3(256), 0, stream>>>(x, base_w, spline_w, spline_s, yws);
    kan_restore_conv<<<dim3(blocks), dim3(256), 0, stream>>>(yws, rw, rb, out);
}
